// GraphConvolutionalGenerator_70463233458736
// MI455X (gfx1250) — compile-verified, matrix operand order checked by reading, not run on hardware
//
#include <hip/hip_runtime.h>

#define B_      256
#define NOISE_  128
#define SEQ_    8
#define G_      12
#define F_      64
#define C_      (G_*F_)            // 768
#define N_NODES (B_*SEQ_*G_)       // 24576
#define NEG_    0.2f
#define EPS_    1e-5f

typedef __attribute__((ext_vector_type(16))) __bf16 v16bf;
typedef __attribute__((ext_vector_type(8)))  float  v8f;

static __device__ __forceinline__ float lrelu(float x) { return x >= 0.f ? x : NEG_ * x; }

// Pack two f32 into two bf16 (round-to-nearest via +0x8000) with one v_perm_b32.
static __device__ __forceinline__ unsigned pack_bf16(float f0, float f1) {
  unsigned x0 = __float_as_uint(f0) + 0x8000u;
  unsigned x1 = __float_as_uint(f1) + 0x8000u;
  return __builtin_amdgcn_perm(x1, x0, 0x07060302u);   // lo16=bf(f0), hi16=bf(f1)
}

// ---------------------------------------------------------------------------
// Generic strided GEMM with up to 2 fused K-segments (for deconv tap pairs):
//   C[c_off + m*c_rs + n*c_cs] = sum_{s<npair} sum_k A[a_off_s + m*a_rs + k*a_cs]
//                                             * B[b_off_s + k*b_rs + n*b_cs]
// bf16 WMMA 16x16x32, f32 accumulate. Block tile 128x64, K-step 32, 8 waves.
// Double-buffered LDS + software pipeline (raw-f32 prefetch, deferred bf16 pack);
// pipeline manually unrolled x2 so the LDS buffer index is a compile-time constant
// (total step count npair*K/32 is even for every GEMM in this net).
// LDS pad stride 20 u32: fragment chunks stay 16B-aligned (ds_load_b128) and the
// 16 rows map to 16 distinct bank groups.  Indices all fit in int32.
// ---------------------------------------------------------------------------
#define TM 128
#define TN 64
#define TK 32
#define LDP 20   // padded u32 row stride (16 kpairs + 4)

__global__ __launch_bounds__(256) void gemm_wmma_bf16(
    const float* __restrict__ A, int a_off0, int a_off1, int a_rs, int a_cs,
    const float* __restrict__ Bm, int b_off0, int b_off1, int b_rs, int b_cs,
    float* __restrict__ Cc, int c_off, int c_rs, int c_cs,
    int N, int K, int npair)
{
  __shared__ __align__(16) unsigned As[2][TM * LDP];   // 2 x 10 KB
  __shared__ __align__(16) unsigned Bs[2][TN * LDP];   // 2 x  5 KB

  const int tid  = threadIdx.x;
  const int lane = tid & 31;
  const int wv   = tid >> 5;              // 0..7
  const int r    = lane & 15, h = lane >> 4;

  const int nblk   = N / TN;
  const int m_base = (blockIdx.x / nblk) * TM;
  const int n_base = (blockIdx.x % nblk) * TN;

  // ---- per-thread staging constants (computed once; segment offsets added later) ----
  int aG[8], aL[8];
#pragma unroll
  for (int q = 0; q < 8; ++q) {
    int p = q * 256 + tid, m = p >> 4, kp = p & 15;
    aG[q] = (m_base + m) * a_rs + 2 * kp * a_cs;
    aL[q] = m * LDP + kp;
  }
  int bG[4], bL[4];
#pragma unroll
  for (int q = 0; q < 4; ++q) {
    int p = q * 256 + tid, n = p >> 4, kp = p & 15;
    bG[q] = 2 * kp * b_rs + (n_base + n) * b_cs;
    bL[q] = n * LDP + kp;
  }

  v8f acc[4];
#pragma unroll
  for (int nb = 0; nb < 4; ++nb)
#pragma unroll
    for (int v = 0; v < 8; ++v) acc[nb][v] = 0.f;

  const int nk    = K / TK;
  const int total = npair * nk;           // even for all GEMMs in this net

  // ---- software pipeline: raw f32 prefetch registers, pack deferred ----
  float ra[16], rb[8];
  auto issue = [&](int kt) {
    int seg = (kt >= nk) ? 1 : 0;
    int k0  = (kt - (seg ? nk : 0)) * TK;
    int ao  = (seg ? a_off1 : a_off0) + k0 * a_cs;
    int bo  = (seg ? b_off1 : b_off0) + k0 * b_rs;
#pragma unroll
    for (int q = 0; q < 8; ++q) {
      int base = aG[q] + ao;
      ra[2 * q]     = A[base];
      ra[2 * q + 1] = A[base + a_cs];
    }
#pragma unroll
    for (int q = 0; q < 4; ++q) {
      int base = bG[q] + bo;
      rb[2 * q]     = Bm[base];
      rb[2 * q + 1] = Bm[base + b_rs];
    }
  };

  union Frag { v16bf v; uint4 u4[2]; };

  auto step = [&](int kt, int buf) {
    // pack (s_wait_loadcnt lands here) and commit to LDS buffer `buf`
#pragma unroll
    for (int q = 0; q < 8; ++q) As[buf][aL[q]] = pack_bf16(ra[2 * q], ra[2 * q + 1]);
#pragma unroll
    for (int q = 0; q < 4; ++q) Bs[buf][bL[q]] = pack_bf16(rb[2 * q], rb[2 * q + 1]);
    __syncthreads();
    if (kt + 1 < total) issue(kt + 1);    // overlap next tile's loads with compute

    Frag af;
    const unsigned* ap = &As[buf][(wv * 16 + r) * LDP + h * 4];
    af.u4[0] = *(const uint4*)ap;
    af.u4[1] = *(const uint4*)(ap + 8);
#pragma unroll
    for (int nb = 0; nb < 4; ++nb) {
      Frag bfr;
      const unsigned* bp = &Bs[buf][(nb * 16 + r) * LDP + h * 4];
      bfr.u4[0] = *(const uint4*)bp;
      bfr.u4[1] = *(const uint4*)(bp + 8);
      acc[nb] = __builtin_amdgcn_wmma_f32_16x16x32_bf16(false, af.v, false, bfr.v,
                                                        (short)0, acc[nb], false, false);
    }
  };

  issue(0);
  for (int kt = 0; kt < total; kt += 2) {
    step(kt, 0);        // compile-time buffer index
    step(kt + 1, 1);
  }

#pragma unroll
  for (int nb = 0; nb < 4; ++nb)
#pragma unroll
    for (int v = 0; v < 8; ++v)
      Cc[c_off + (m_base + wv * 16 + v + 8 * h) * c_rs +
         (n_base + nb * 16 + r) * c_cs] = acc[nb][v];
}

// ---------------------------------------------------------------------------
// Training-mode BatchNorm (biased var) + LeakyReLU, in place.
// X layout [Bn, Cch, L]; stats per channel over (Bn, L). One thread per channel.
// ---------------------------------------------------------------------------
__global__ void bn_lrelu_kernel(float* __restrict__ X, const float* __restrict__ g,
                                const float* __restrict__ b, int Cch, int Bn, int L)
{
  int c = blockIdx.x * blockDim.x + threadIdx.x;
  if (c >= Cch) return;
  float mean = 0.f, msq = 0.f;
  const float inv = 1.f / (float)(Bn * L);
  for (int i = 0; i < Bn; ++i)
    for (int l = 0; l < L; ++l) {
      float x = X[(long)i * Cch * L + (long)c * L + l];
      mean += x; msq += x * x;
    }
  mean *= inv; msq *= inv;
  float var = msq - mean * mean;
  float sc = g[c] * rsqrtf(var + EPS_);
  float sh = b[c] - mean * sc;
  for (int i = 0; i < Bn; ++i)
    for (int l = 0; l < L; ++l) {
      long idx = (long)i * Cch * L + (long)c * L + l;
      X[idx] = lrelu(X[idx] * sc + sh);
    }
}

// nodes[n,f] = X3[b, g*F+f, l] with n = (b*SEQ + l)*G + g
__global__ void make_nodes_kernel(const float* __restrict__ X3, float* __restrict__ nodes)
{
  int t = blockIdx.x * blockDim.x + threadIdx.x;
  if (t >= N_NODES * F_) return;
  int f = t & (F_ - 1);
  int n = t >> 6;
  int g = n % G_;
  int l = (n / G_) % SEQ_;
  int b = n / (G_ * SEQ_);
  nodes[t] = X3[(long)b * (C_ * SEQ_) + (long)(g * F_ + f) * SEQ_ + l];
}

__global__ void attn_scores_kernel(const float* __restrict__ xp, const float* __restrict__ a_src,
                                   const float* __restrict__ a_dst, float* __restrict__ as_,
                                   float* __restrict__ ad_)
{
  int n = blockIdx.x * blockDim.x + threadIdx.x;
  if (n >= N_NODES) return;
  float s0 = 0.f, s1 = 0.f;
  for (int f = 0; f < F_; ++f) {
    float v = xp[(long)n * F_ + f];
    s0 += v * a_src[f];
    s1 += v * a_dst[f];
  }
  as_[n] = s0; ad_[n] = s1;
}

__global__ void init_kernel(float* __restrict__ m, float* __restrict__ s,
                            float* __restrict__ out)
{
  int t = blockIdx.x * blockDim.x + threadIdx.x;
  if (t < N_NODES) { m[t] = -__builtin_inff(); s[t] = 0.f; }
  if (t < N_NODES * F_) out[t] = 0.f;
}

static __device__ __forceinline__ void edge_sd(const int* __restrict__ ei, int E, int e,
                                               int& src, int& dst)
{
  if (e < E) { src = ei[e]; dst = ei[E + e]; }
  else       { src = dst = e - E; }            // self loops appended
}

__global__ void edge_max_kernel(const int* __restrict__ ei, int E, int ET,
                                const float* __restrict__ as_, const float* __restrict__ ad_,
                                float* __restrict__ m)
{
  int e = blockIdx.x * blockDim.x + threadIdx.x;
  if (e >= ET) return;
  int src, dst; edge_sd(ei, E, e, src, dst);
  float ev = lrelu(as_[src] + ad_[dst]);
  // float atomic-max via ordered-int trick (m initialized to -inf)
  if (ev >= 0.f) atomicMax((int*)(m + dst), __float_as_int(ev));
  else           atomicMin((unsigned int*)(m + dst), (unsigned int)__float_as_int(ev));
}

__global__ void edge_sum_kernel(const int* __restrict__ ei, int E, int ET,
                                const float* __restrict__ as_, const float* __restrict__ ad_,
                                const float* __restrict__ m, float* __restrict__ s)
{
  int e = blockIdx.x * blockDim.x + threadIdx.x;
  if (e >= ET) return;
  int src, dst; edge_sd(ei, E, e, src, dst);
  float ev = lrelu(as_[src] + ad_[dst]);
  atomicAdd(s + dst, __expf(ev - m[dst]));
}

// One wave per edge, 2 features per lane (F=64, wave32).
__global__ void edge_scatter_kernel(const int* __restrict__ ei, int E, int ET,
                                    const float* __restrict__ as_, const float* __restrict__ ad_,
                                    const float* __restrict__ m, const float* __restrict__ s,
                                    const float* __restrict__ xp, float* __restrict__ out)
{
  int lane = threadIdx.x & 31;
  int e = blockIdx.x * (blockDim.x >> 5) + (threadIdx.x >> 5);
  if (e >= ET) return;
  int src, dst; edge_sd(ei, E, e, src, dst);
  float ev = lrelu(as_[src] + ad_[dst]);
  float alpha = __expf(ev - m[dst]) / s[dst];
  atomicAdd(out + (long)dst * F_ + lane,      alpha * xp[(long)src * F_ + lane]);
  atomicAdd(out + (long)dst * F_ + lane + 32, alpha * xp[(long)src * F_ + lane + 32]);
}

__global__ void final_kernel(float* __restrict__ out, const float* __restrict__ bg)
{
  int t = blockIdx.x * blockDim.x + threadIdx.x;
  if (t >= N_NODES * F_) return;
  out[t] = tanhf(out[t] + bg[t & (F_ - 1)]);
}

// ---------------------------------------------------------------------------
extern "C" void kernel_launch(void* const* d_in, const int* in_sizes, int n_in,
                              void* d_out, int out_size, void* d_ws, size_t ws_size,
                              hipStream_t stream)
{
  (void)n_in; (void)out_size; (void)ws_size;
  const float* noise      = (const float*)d_in[0];
  const int*   edge_index = (const int*)d_in[1];
  /* d_in[2] edge_weight: ignored by GATConv per reference */
  const float* W_lin = (const float*)d_in[3];
  const float* g0 = (const float*)d_in[4];
  const float* b0 = (const float*)d_in[5];
  const float* w1 = (const float*)d_in[6];
  const float* g1 = (const float*)d_in[7];
  const float* b1 = (const float*)d_in[8];
  const float* w2 = (const float*)d_in[9];
  const float* g2 = (const float*)d_in[10];
  const float* b2 = (const float*)d_in[11];
  const float* w3 = (const float*)d_in[12];
  const float* W_gat = (const float*)d_in[13];
  const float* a_src = (const float*)d_in[14];
  const float* a_dst = (const float*)d_in[15];
  const float* b_gat = (const float*)d_in[16];

  float* out = (float*)d_out;
  float* ws  = (float*)d_ws;
  const long SZ = (long)B_ * 8 * C_;   // 1,572,864 floats; also == N_NODES*F_
  float* H     = ws;
  float* X1    = H + SZ;
  float* X2    = X1 + SZ;
  float* X3    = X2 + SZ;
  float* nodes = X3 + SZ;
  float* xp    = nodes + SZ;
  float* as_   = xp + SZ;
  float* ad_   = as_ + N_NODES;
  float* mx    = ad_ + N_NODES;
  float* sm    = mx + N_NODES;

  const int E  = in_sizes[1] / 2;
  const int ET = E + N_NODES;

  auto gemm = [&](const float* A, int ao0, int ao1, int ars, int acs,
                  const float* Bm, int bo0, int bo1, int brs, int bcs,
                  float* Cc, int co, int crs, int ccs,
                  int M, int N, int K, int npair) {
    int blocks = (M / TM) * (N / TN);
    gemm_wmma_bf16<<<blocks, 256, 0, stream>>>(A, ao0, ao1, ars, acs,
                                               Bm, bo0, bo1, brs, bcs,
                                               Cc, co, crs, ccs, N, K, npair);
  };

  // ---- Linear: H[B,8C] = noise @ W_lin; BN(axis 0) + LeakyReLU ----
  gemm(noise, 0, 0, NOISE_, 1, W_lin, 0, 0, 8 * C_, 1, H, 0, 8 * C_, 1,
       B_, 8 * C_, NOISE_, 1);
  bn_lrelu_kernel<<<(8 * C_ + 255) / 256, 256, 0, stream>>>(H, g0, b0, 8 * C_, B_, 1);

  // ---- Deconv1: [B,8C,1] -> [B,4C,2]; single tap t=j+1 per output ----
  for (int j = 0; j < 2; ++j)
    gemm(H, 0, 0, 8 * C_, 1,
         w1, j + 1, 0, 4 * C_ * 4, 4,
         X1, j, 4 * C_ * 2, 2,
         B_, 4 * C_, 8 * C_, 1);
  bn_lrelu_kernel<<<(4 * C_ + 255) / 256, 256, 0, stream>>>(X1, g1, b1, 4 * C_, B_, 2);

  // ---- Deconv2: [B,4C,2] -> [B,2C,4]; fuse the <=2 (i,t) pairs per j ----
  for (int j = 0; j < 4; ++j) {
    int ao[2] = {0, 0}, bo[2] = {0, 0}, np = 0;
    for (int i = 0; i < 2; ++i) {
      int t = j - 2 * i + 1;
      if (t < 0 || t > 3) continue;
      ao[np] = i; bo[np] = t; ++np;
    }
    gemm(X1, ao[0], ao[1], 4 * C_ * 2, 2,
         w2, bo[0], bo[1], 2 * C_ * 4, 4,
         X2, j, 2 * C_ * 4, 4,
         B_, 2 * C_, 4 * C_, np);
  }
  bn_lrelu_kernel<<<(2 * C_ + 255) / 256, 256, 0, stream>>>(X2, g2, b2, 2 * C_, B_, 4);

  // ---- Deconv3: [B,2C,4] -> [B,C,8]; fuse pairs; no BN/activation ----
  for (int j = 0; j < 8; ++j) {
    int ao[2] = {0, 0}, bo[2] = {0, 0}, np = 0;
    for (int i = 0; i < 4; ++i) {
      int t = j - 2 * i + 1;
      if (t < 0 || t > 3) continue;
      ao[np] = i; bo[np] = t; ++np;
    }
    gemm(X2, ao[0], ao[1], 2 * C_ * 4, 4,
         w3, bo[0], bo[1], C_ * 4, 4,
         X3, j, C_ * 8, 8,
         B_, C_, 2 * C_, np);
  }

  // ---- GAT ----
  make_nodes_kernel<<<(N_NODES * F_ + 255) / 256, 256, 0, stream>>>(X3, nodes);
  gemm(nodes, 0, 0, F_, 1, W_gat, 0, 0, F_, 1, xp, 0, F_, 1, N_NODES, F_, F_, 1);
  attn_scores_kernel<<<(N_NODES + 255) / 256, 256, 0, stream>>>(xp, a_src, a_dst, as_, ad_);
  init_kernel<<<(N_NODES * F_ + 255) / 256, 256, 0, stream>>>(mx, sm, out);
  edge_max_kernel<<<(ET + 255) / 256, 256, 0, stream>>>(edge_index, E, ET, as_, ad_, mx);
  edge_sum_kernel<<<(ET + 255) / 256, 256, 0, stream>>>(edge_index, E, ET, as_, ad_, mx, sm);
  edge_scatter_kernel<<<(ET + 7) / 8, 256, 0, stream>>>(edge_index, E, ET, as_, ad_, mx, sm,
                                                        xp, out);
  final_kernel<<<(N_NODES * F_ + 255) / 256, 256, 0, stream>>>(out, b_gat);
}
